// MultiBandGATWithNodeImportance_8899172237589
// MI455X (gfx1250) — matrix operand order repaired and checked by reading, not compiled, not run on hardware
//
#include <hip/hip_runtime.h>
#include <hip/hip_bf16.h>

// ---------------------------------------------------------------------------
// MultiBandGAT forward for gfx1250 (MI455X).
// GEMMs on the WMMA pipe (v_wmma_f32_16x16x32_f16, f16 in / f32 acc).
// B operands (and the whole per-graph H in the Gram kernel) are staged in LDS
// as f16 with bank padding, so WMMA fragments come from ds_load_b128s.
// GAT biases dropped (BN-train is invariant to per-channel constant shifts).
// Edge softmax: monotone-uint atomicMax + atomicAdd scatter.
// ---------------------------------------------------------------------------

typedef __attribute__((ext_vector_type(16))) _Float16 v16h;
typedef __attribute__((ext_vector_type(8)))  _Float16 v8h;
typedef __attribute__((ext_vector_type(8)))  float    v8f;

#define NN    32768          // total nodes
#define GG    128            // graphs
#define NPG   256            // nodes per graph
#define HID   64
#define EDG   524288         // edges per band (G * NPG * DEG)
#define ETOT  (EDG + NN)     // + self loops
#define ADJSZ ((size_t)GG * NPG * NPG)
#define HS_STRIDE 72         // LDS row stride (halfs) for H staging (bank pad)

// ---------------- WMMA fragment loaders (ISA 7.12.2 lane layouts) ----------

// A (16x32 f16) from global fp32, vectorized float4 reads.
// lanes 0-15 -> M=lane, K in {kb..kb+7, kb+16..kb+23}, kb=0; lanes 16-31 kb=8.
__device__ __forceinline__ v16h load_a_frag_g(const float* __restrict__ A,
                                              int lda, int mBase, int k0, int lane) {
  int m  = mBase + (lane & 15);
  int kb = k0 + ((lane & 16) ? 8 : 0);
  const float4* p = (const float4*)(A + (size_t)m * lda + kb);
  float4 u0 = p[0], u1 = p[1], u2 = p[4], u3 = p[5];
  v16h a;
  a[0]  = (_Float16)u0.x; a[1]  = (_Float16)u0.y; a[2]  = (_Float16)u0.z; a[3]  = (_Float16)u0.w;
  a[4]  = (_Float16)u1.x; a[5]  = (_Float16)u1.y; a[6]  = (_Float16)u1.z; a[7]  = (_Float16)u1.w;
  a[8]  = (_Float16)u2.x; a[9]  = (_Float16)u2.y; a[10] = (_Float16)u2.z; a[11] = (_Float16)u2.w;
  a[12] = (_Float16)u3.x; a[13] = (_Float16)u3.y; a[14] = (_Float16)u3.z; a[15] = (_Float16)u3.w;
  return a;
}

// A fragment from LDS-staged f16 rows (row stride in halfs).
__device__ __forceinline__ v16h lds_a_frag(const _Float16* Hs, int stride,
                                           int mBase, int k0, int lane) {
  int row = mBase + (lane & 15);
  int kb  = k0 + ((lane & 16) ? 8 : 0);
  const _Float16* p = Hs + row * stride + kb;
  v8h lo = *(const v8h*)p;
  v8h hi = *(const v8h*)(p + 16);
  v16h a;
#pragma unroll
  for (int e = 0; e < 8; ++e) { a[e] = lo[e]; a[8 + e] = hi[e]; }
  return a;
}

// B fragment (32x16): lane holds column n, K in {kb..kb+7, kb+16..kb+23}, kb=0 (lanes 0-15) / 8 (lanes 16-31).
// Source rows store B^T (i.e. row n holds B[.,n] over k) so reads are contiguous.
__device__ __forceinline__ v16h lds_b_frag(const _Float16* Bs, int stride,
                                           int nBase, int k0, int lane) {
  int row = nBase + (lane & 15);
  int kb  = k0 + ((lane & 16) ? 8 : 0);
  const _Float16* p = Bs + row * stride + kb;
  v8h lo = *(const v8h*)p;
  v8h hi = *(const v8h*)(p + 16);
  v16h b;
#pragma unroll
  for (int e = 0; e < 8; ++e) { b[e] = lo[e]; b[8 + e] = hi[e]; }
  return b;
}

// ---------------- GEMM: C[M,64] = act(A[M,K] @ B[K,64] + bias) -------------
// blockDim 256 (8 waves = 2 m-tiles x 4 n-tiles => 32 rows/block).
// B staged transposed in LDS once per block.
__global__ void gemm_wmma_kernel(const float* __restrict__ A,
                                 const float* __restrict__ B,
                                 float* __restrict__ C,
                                 const float* __restrict__ bias,
                                 int K, int act) {
  __shared__ _Float16 Bs[64 * 200];           // stride K+8, K<=192
  int t = threadIdx.x, lane = t & 31, wave = t >> 5;
  int stride = K + 8;
  for (int i = t; i < K * 64; i += 256) {     // B row-major [K][64] -> Bs[n][k]
    int k = i >> 6, n = i & 63;
    Bs[n * stride + k] = (_Float16)B[i];
  }
  __syncthreads();

  int mBase = blockIdx.x * 32 + (wave & 1) * 16;
  int nBase = (wave >> 1) * 16;
  v8f acc = {};
  for (int k0 = 0; k0 < K; k0 += 32) {
    __builtin_prefetch(A + (size_t)(mBase + (lane & 15)) * K + k0 + 64, 0, 0);
    v16h a = load_a_frag_g(A, K, mBase, k0, lane);
    v16h b = lds_b_frag(Bs, stride, nBase, k0, lane);
    acc = __builtin_amdgcn_wmma_f32_16x16x32_f16(false, a, false, b,
                                                 (short)0, acc, false, false);
  }
  int   n    = nBase + (lane & 15);
  float bv   = bias ? bias[n] : 0.0f;
  int   moff = (lane & 16) ? 8 : 0;
#pragma unroll
  for (int r = 0; r < 8; ++r) {
    float v = acc[r] + bv;
    if (act == 1) v = fmaxf(v, 0.0f);
    C[(size_t)(mBase + moff + r) * HID + n] = v;
  }
}

// ---------------- Per-graph Gram: out = sigmoid(H H^T) --------------------
// grid (NPG/16, G), blockDim 256 (8 waves). Whole H_g staged once in LDS (f16,
// padded stride); both A and B fragments come from LDS. Each wave: 2 n-tiles.
__global__ void gram_sigmoid_kernel(const float* __restrict__ H,
                                    float* __restrict__ Out) {
  __shared__ _Float16 Hs[NPG * HS_STRIDE];    // 36 KB
  int t = threadIdx.x, lane = t & 31, wave = t >> 5;
  int g = blockIdx.y;
  const float* Hg = H + (size_t)g * NPG * HID;
  for (int i = t * 4; i < NPG * HID; i += 256 * 4) {
    float4 v = *(const float4*)(Hg + i);
    _Float16* q = Hs + (i >> 6) * HS_STRIDE + (i & 63);
    q[0] = (_Float16)v.x; q[1] = (_Float16)v.y;
    q[2] = (_Float16)v.z; q[3] = (_Float16)v.w;
  }
  __syncthreads();

  int    mBase = blockIdx.x * 16;
  int    moff  = (lane & 16) ? 8 : 0;
  float* outg  = Out + (size_t)g * NPG * NPG;
  for (int tt = wave; tt < 16; tt += 8) {
    int nBase = tt * 16;
    v8f acc = {};
#pragma unroll
    for (int k0 = 0; k0 < HID; k0 += 32) {
      v16h a = lds_a_frag(Hs, HS_STRIDE, mBase, k0, lane);
      v16h b = lds_b_frag(Hs, HS_STRIDE, nBase, k0, lane);
      acc = __builtin_amdgcn_wmma_f32_16x16x32_f16(false, a, false, b,
                                                   (short)0, acc, false, false);
    }
    int n = nBase + (lane & 15);
#pragma unroll
    for (int r = 0; r < 8; ++r) {
      float v = acc[r];
      outg[(size_t)(mBase + moff + r) * NPG + n] = 1.0f / (1.0f + __expf(-v));
    }
  }
}

// ---------------- GAT attention / segment softmax -------------------------

__device__ __forceinline__ unsigned fkey(float f) {
  unsigned u = __float_as_uint(f);
  return (u & 0x80000000u) ? ~u : (u | 0x80000000u);
}
__device__ __forceinline__ float fval(unsigned k) {
  return (k & 0x80000000u) ? __uint_as_float(k & 0x7FFFFFFFu)
                           : __uint_as_float(~k);
}

__global__ void attn_scores_kernel(const float* __restrict__ h,
                                   const float* __restrict__ att_src,
                                   const float* __restrict__ att_dst,
                                   float* __restrict__ a_s,
                                   float* __restrict__ a_d) {
  int i = blockIdx.x * blockDim.x + threadIdx.x;
  if (i >= NN) return;
  const float* row = h + (size_t)i * HID;
  float s = 0.f, d = 0.f;
#pragma unroll 8
  for (int c = 0; c < HID; ++c) { s += row[c] * att_src[c]; d += row[c] * att_dst[c]; }
  a_s[i] = s; a_d[i] = d;
}

__global__ void band_init_kernel(float* __restrict__ gatout,
                                 float* __restrict__ ssum,
                                 unsigned* __restrict__ mkey,
                                 float* __restrict__ stats) {
  int i = blockIdx.x * blockDim.x + threadIdx.x;
  if (i < NN * HID) gatout[i] = 0.f;
  if (i < NN) { ssum[i] = 0.f; mkey[i] = 0u; }   // key(0) < key(any finite)
  if (i < 128) stats[i] = 0.f;
}

__device__ __forceinline__ void edge_pair(const int* __restrict__ ei, int e,
                                          int& src, int& dst) {
  if (e < EDG) { src = ei[e]; dst = ei[EDG + e]; }
  else         { src = dst = e - EDG; }          // self loop
}

__global__ void edge_max_kernel(const int* __restrict__ ei,
                                const float* __restrict__ a_s,
                                const float* __restrict__ a_d,
                                unsigned* __restrict__ mkey) {
  int e = blockIdx.x * blockDim.x + threadIdx.x;
  if (e >= ETOT) return;
  int src, dst; edge_pair(ei, e, src, dst);
  float l = a_s[src] + a_d[dst];
  l = (l > 0.f) ? l : 0.2f * l;                  // leaky_relu 0.2
  atomicMax(mkey + dst, fkey(l));
}

__global__ void edge_expsum_kernel(const int* __restrict__ ei,
                                   const float* __restrict__ a_s,
                                   const float* __restrict__ a_d,
                                   const unsigned* __restrict__ mkey,
                                   float* __restrict__ ssum,
                                   float* __restrict__ ebuf) {
  int e = blockIdx.x * blockDim.x + threadIdx.x;
  if (e >= ETOT) return;
  int src, dst; edge_pair(ei, e, src, dst);
  float l = a_s[src] + a_d[dst];
  l = (l > 0.f) ? l : 0.2f * l;
  float ex = __expf(l - fval(mkey[dst]));
  ebuf[e] = ex;
  atomicAdd(ssum + dst, ex);
}

// one thread = one edge x 4 channels  (ETOT*16 threads)
__global__ void edge_agg_kernel(const int* __restrict__ ei,
                                const float* __restrict__ h,
                                const float* __restrict__ ssum,
                                const float* __restrict__ ebuf,
                                float* __restrict__ gatout) {
  int tid = blockIdx.x * blockDim.x + threadIdx.x;
  if (tid >= ETOT * 16) return;
  int e  = tid >> 4;
  int c0 = (tid & 15) * 4;
  int src, dst; edge_pair(ei, e, src, dst);
  float alpha = ebuf[e] / ssum[dst];
  const float* hs = h + (size_t)src * HID + c0;
  float*       o  = gatout + (size_t)dst * HID + c0;
#pragma unroll
  for (int j = 0; j < 4; ++j) atomicAdd(o + j, hs[j] * alpha);
}

// ---------------- BatchNorm (training stats) + ReLU -----------------------
// blockDim 256, 256 rows per block, N/256 blocks
__global__ void bn_stats_kernel(const float* __restrict__ x,
                                float* __restrict__ stats) {
  __shared__ float sA[256];
  __shared__ float sB[256];
  int t    = threadIdx.x;
  int c    = t & 63;
  int p    = t >> 6;                 // 0..3
  int base = blockIdx.x * 256;
  float s1 = 0.f, s2 = 0.f;
  for (int r = p; r < 256; r += 4) {
    float v = x[(size_t)(base + r) * HID + c];
    s1 += v; s2 += v * v;
  }
  sA[t] = s1; sB[t] = s2;
  __syncthreads();
  if (t < 64) {
    float a = sA[t] + sA[64 + t] + sA[128 + t] + sA[192 + t];
    float b = sB[t] + sB[64 + t] + sB[128 + t] + sB[192 + t];
    atomicAdd(stats + t,      a);
    atomicAdd(stats + 64 + t, b);
  }
}

__global__ void bn_apply_kernel(const float* __restrict__ x,
                                const float* __restrict__ stats,
                                const float* __restrict__ gamma,
                                const float* __restrict__ beta,
                                float* __restrict__ dst,
                                int dstld, int colOff) {
  int idx = blockIdx.x * blockDim.x + threadIdx.x;
  if (idx >= NN * HID) return;
  int   row = idx >> 6;
  int   c   = idx & 63;
  float mu  = stats[c] * (1.0f / NN);
  float var = stats[64 + c] * (1.0f / NN) - mu * mu;
  float y   = gamma[c] * (x[idx] - mu) * rsqrtf(var + 1e-5f) + beta[c];
  dst[(size_t)row * dstld + colOff + c] = fmaxf(y, 0.f);
}

// ---------------- classifier head -----------------------------------------

__global__ void pool_kernel(const float* __restrict__ z,
                            const float* __restrict__ gf,
                            float* __restrict__ xcls) {
  int idx = blockIdx.x * blockDim.x + threadIdx.x;
  if (idx >= GG * 71) return;
  int g = idx / 71, c = idx % 71;
  float v;
  if (c < 64) {
    float s = 0.f;
    const float* p = z + (size_t)g * NPG * HID + c;
    for (int r = 0; r < NPG; ++r) s += p[(size_t)r * HID];
    v = s * (1.0f / NPG);
  } else {
    v = gf[g * 7 + (c - 64)];
  }
  xcls[idx] = v;
}

// one 64-thread block per graph
__global__ void classifier_kernel(const float* __restrict__ xcls,
                                  const float* __restrict__ W1, const float* __restrict__ b1,
                                  const float* __restrict__ W2, const float* __restrict__ b2,
                                  const float* __restrict__ W3, const float* __restrict__ b3,
                                  const float* __restrict__ W4, const float* __restrict__ b4,
                                  float* __restrict__ logits) {
  __shared__ float xin[71];
  __shared__ float l1[64];
  __shared__ float l2[16];
  __shared__ float l3[8];
  int g = blockIdx.x, t = threadIdx.x;
  for (int i = t; i < 71; i += 64) xin[i] = xcls[g * 71 + i];
  __syncthreads();
  float acc = b1[t];
  for (int i = 0; i < 71; ++i) acc += xin[i] * W1[i * 64 + t];
  l1[t] = fmaxf(acc, 0.f);
  __syncthreads();
  if (t < 16) { float a = b2[t]; for (int i = 0; i < 64; ++i) a += l1[i] * W2[i * 16 + t]; l2[t] = fmaxf(a, 0.f); }
  __syncthreads();
  if (t < 8)  { float a = b3[t]; for (int i = 0; i < 16; ++i) a += l2[i] * W3[i * 8 + t];  l3[t] = fmaxf(a, 0.f); }
  __syncthreads();
  if (t == 0) { float a = b4[0]; for (int i = 0; i < 8; ++i)  a += l3[i] * W4[i]; logits[g] = a; }
}

// ---------------------------------------------------------------------------

extern "C" void kernel_launch(void* const* d_in, const int* in_sizes, int n_in,
                              void* d_out, int out_size, void* d_ws, size_t ws_size,
                              hipStream_t stream) {
  auto F = [&](int i) { return (const float*)d_in[i]; };
  auto I = [&](int i) { return (const int*)d_in[i]; };

  // pytree(sorted-keys) flattening of params, then remaining dict entries
  const float *bn_b_[5] = {F(0), F(2), F(4), F(6), F(8)};   // bn_a,bn_b,bn_f,bn_o,bn_t : b
  const float *bn_g_[5] = {F(1), F(3), F(5), F(7), F(9)};   //                          : g
  const float *clsW1 = F(10), *clsW2 = F(11), *clsW3 = F(12), *clsW4 = F(13);
  const float *clsb1 = F(14), *clsb2 = F(15), *clsb3 = F(16), *clsb4 = F(17);
  const float *decW1[3] = {F(18), F(22), F(26)};
  const float *decW2[3] = {F(19), F(23), F(27)};
  const float *decb1[3] = {F(20), F(24), F(28)};
  const float *decb2[3] = {F(21), F(25), F(29)};
  // gat_a, gat_b, gat_f, gat_o, gat_t : {W, att_dst, att_src, b}
  const float *gatW[5]  = {F(30), F(34), F(38), F(42), F(46)};
  const float *gatAd[5] = {F(31), F(35), F(39), F(43), F(47)};
  const float *gatAs[5] = {F(32), F(36), F(40), F(44), F(48)};
  const float *x_band[3] = {F(50), F(51), F(52)};
  const float *gfeat = F(53);
  const int   *ei_band[3] = {I(54), I(55), I(56)};

  // ---- workspace layout (floats) ----
  float* ws = (float*)d_ws;
  size_t off = 0;
  auto alloc = [&](size_t n) { float* p = ws + off; off += n; return p; };
  float*    h      = alloc((size_t)NN * HID);
  float*    gatout = alloc((size_t)NN * HID);
  float*    a_s    = alloc(NN);
  float*    a_d    = alloc(NN);
  float*    ssum   = alloc(NN);
  unsigned* mkey   = (unsigned*)alloc(NN);
  float*    ebuf   = alloc(ETOT);
  float*    stats  = alloc(128);
  float*    zc     = alloc((size_t)NN * 192);
  float*    zf     = alloc((size_t)NN * HID);
  float*    z      = alloc((size_t)NN * HID);
  float*    h1     = alloc((size_t)NN * HID);
  float*    h2     = alloc((size_t)NN * HID);
  float*    xcls   = alloc((size_t)GG * 71);

  float* out = (float*)d_out;

  dim3 blk256(256), blk64(64);
  dim3 gridNH(((NN * HID) + 255) / 256);
  dim3 gridE((ETOT + 255) / 256);
  dim3 gridEA(((size_t)ETOT * 16 + 255) / 256);

  // one full GATConv + BN-ReLU band (bias dropped: BN-train invariant)
  auto run_gat = [&](const float* x, int K, int gi, int bni, const int* ei,
                     float* dstbuf, int dstld, int colOff) {
    gemm_wmma_kernel<<<dim3(NN / 32), blk256, 0, stream>>>(
        x, gatW[gi], h, nullptr, K, 0);
    attn_scores_kernel<<<dim3((NN + 255) / 256), blk256, 0, stream>>>(
        h, gatAs[gi], gatAd[gi], a_s, a_d);
    band_init_kernel<<<gridNH, blk256, 0, stream>>>(gatout, ssum, mkey, stats);
    edge_max_kernel<<<gridE, blk256, 0, stream>>>(ei, a_s, a_d, mkey);
    edge_expsum_kernel<<<gridE, blk256, 0, stream>>>(ei, a_s, a_d, mkey, ssum, ebuf);
    edge_agg_kernel<<<gridEA, blk256, 0, stream>>>(ei, h, ssum, ebuf, gatout);
    bn_stats_kernel<<<dim3(NN / 256), blk256, 0, stream>>>(gatout, stats);
    bn_apply_kernel<<<gridNH, blk256, 0, stream>>>(
        gatout, stats, bn_g_[bni], bn_b_[bni], dstbuf, dstld, colOff);
  };

  // --- encoder: three bands -> zc [N,192] ---
  run_gat(x_band[0], 128, 0, 0, ei_band[0], zc, 192, 0);     // alpha
  run_gat(x_band[1], 128, 1, 1, ei_band[1], zc, 192, 64);    // beta
  run_gat(x_band[2], 128, 4, 4, ei_band[2], zc, 192, 128);   // theta (gat_t, bn_t)
  // fusion + output conv (both on alpha edges)
  run_gat(zc, 192, 2, 2, ei_band[0], zf, HID, 0);            // gat_f, bn_f
  run_gat(zf, 64, 3, 3, ei_band[0], z, HID, 0);              // gat_o, bn_o

  // --- decoder: per band, h = relu(relu(z W1 + b1) W2 + b2); sigmoid(h h^T)
  for (int b = 0; b < 3; ++b) {
    gemm_wmma_kernel<<<dim3(NN / 32), blk256, 0, stream>>>(
        z, decW1[b], h1, decb1[b], HID, 1);
    gemm_wmma_kernel<<<dim3(NN / 32), blk256, 0, stream>>>(
        h1, decW2[b], h2, decb2[b], HID, 1);
    gram_sigmoid_kernel<<<dim3(NPG / 16, GG), blk256, 0, stream>>>(
        h2, out + (size_t)b * ADJSZ);
  }

  // --- classifier ---
  pool_kernel<<<dim3((GG * 71 + 255) / 256), blk256, 0, stream>>>(z, gfeat, xcls);
  classifier_kernel<<<dim3(GG), blk64, 0, stream>>>(
      xcls, clsW1, clsb1, clsW2, clsb2, clsW3, clsb3, clsW4, clsb4,
      out + 3 * ADJSZ);
}
